// ResidualGAT_6906307412650
// MI455X (gfx1250) — compile-verified, matrix operand order checked by reading, not run on hardware
//
#include <hip/hip_runtime.h>

// ---------------------------------------------------------------------------
// CDNA5 (gfx1250) WMMA types + async global->LDS helpers
// ---------------------------------------------------------------------------
typedef __attribute__((ext_vector_type(16))) __bf16 v16bf;
typedef __attribute__((ext_vector_type(8)))  float  v8f;
typedef __attribute__((ext_vector_type(2)))  int    i32x2;
typedef __attribute__((ext_vector_type(4)))  int    i32x4;

#define HH 128            // hidden per head
#define CC 256            // HEADS*H

#if defined(__AMDGCN__) && \
    __has_builtin(__builtin_amdgcn_global_load_async_to_lds_b64) && \
    __has_builtin(__builtin_amdgcn_global_load_async_to_lds_b128)
#define HAVE_ASYNC 1
#else
#define HAVE_ASYNC 0
#endif

#if HAVE_ASYNC
__device__ __forceinline__ void async_b64(void* l, const void* g) {
    __builtin_amdgcn_global_load_async_to_lds_b64(
        (__attribute__((address_space(1))) i32x2*)g,
        (__attribute__((address_space(3))) i32x2*)l, 0, 0);
}
__device__ __forceinline__ void async_b128(void* l, const void* g) {
    __builtin_amdgcn_global_load_async_to_lds_b128(
        (__attribute__((address_space(1))) i32x4*)g,
        (__attribute__((address_space(3))) i32x4*)l, 0, 0);
}
#endif

__device__ __forceinline__ void wait_async_lds() {
#if HAVE_ASYNC
#if __has_builtin(__builtin_amdgcn_s_wait_asynccnt)
    __builtin_amdgcn_s_wait_asynccnt(0);
#else
    asm volatile("s_wait_asynccnt 0x0" ::: "memory");
#endif
#endif
}

__device__ __forceinline__ unsigned encodeOrdered(float v) {
    unsigned b = __float_as_uint(v);
    return (b & 0x80000000u) ? ~b : (b | 0x80000000u);
}
__device__ __forceinline__ float decodeOrdered(unsigned e) {
    unsigned b = (e & 0x80000000u) ? (e & 0x7fffffffu) : ~e;
    float f = __uint_as_float(b);
    // non-finite (incl. the all-zero init pattern) -> 0, matching jnp.where(isfinite)
    return ((b & 0x7f800000u) == 0x7f800000u) ? 0.0f : f;
}

// ---------------------------------------------------------------------------
// Weight prep: transpose + bf16 convert all GEMM weights.
//  wct[col][k] : EdgeConv fused W (F x 2H), col<128 -> W1a-W1b, col>=128 -> W1b
//  w2t[col][k], g1t[col][k], g2t[col][k] : plain transposes
// ---------------------------------------------------------------------------
__global__ void prep_weights_k(const float* __restrict__ ecW1,
                               const float* __restrict__ ecW2,
                               const float* __restrict__ g1W,
                               const float* __restrict__ g2W,
                               __bf16* __restrict__ wct,
                               __bf16* __restrict__ w2t,
                               __bf16* __restrict__ g1t,
                               __bf16* __restrict__ g2t) {
    int i = blockIdx.x * 256 + threadIdx.x;
    if (i < 32768) {                       // wct: 256 cols x 128 k
        int col = i >> 7, k = i & 127;
        float v;
        if (col < 128) v = ecW1[k * 128 + col] - ecW1[(128 + k) * 128 + col];
        else           v = ecW1[(128 + k) * 128 + (col - 128)];
        wct[col * 128 + k] = (__bf16)v;
    } else if (i < 49152) {                // w2t: 128x128
        int j = i - 32768;
        int col = j >> 7, k = j & 127;
        w2t[col * 128 + k] = (__bf16)ecW2[k * 128 + col];
    } else if (i < 81920) {                // g1t: 256 cols x 128 k
        int j = i - 49152;
        int col = j >> 7, k = j & 127;
        g1t[col * 128 + k] = (__bf16)g1W[k * 256 + col];
    } else if (i < 147456) {               // g2t: 256 cols x 256 k
        int j = i - 81920;
        int col = j >> 8, k = j & 255;
        g2t[col * 256 + k] = (__bf16)g2W[k * 256 + col];
    }
}

__global__ void convert_bf_k(const float* __restrict__ in, __bf16* __restrict__ out,
                             long count) {
    long i = (long)blockIdx.x * 256 + threadIdx.x;
    if (i < count) out[i] = (__bf16)in[i];
}

// ---------------------------------------------------------------------------
// Generic bf16 WMMA GEMM: C[M x ncols] = A[M x K] @ Bt[ncols x K]^T  (fp32 acc)
// Block: 256 thr / 8 waves; block tile 32 rows x 64 cols; wave -> one 16x16.
// Double-buffered LDS staging via GLOBAL_LOAD_ASYNC_TO_LDS (ASYNCcnt).
// ---------------------------------------------------------------------------
__global__ void gemm_bf16_k(const __bf16* __restrict__ A,
                            const __bf16* __restrict__ Bt,
                            float* __restrict__ C,
                            int M, int K, int ncols) {
    __shared__ __bf16 As[2][32][32];
    __shared__ __bf16 Bs[2][64][32];
    const int tid  = threadIdx.x;
    const int lane = tid & 31;
    const int wave = tid >> 5;
    const int g    = lane >> 4;      // half-wave group
    const int rc   = lane & 15;      // row (A) / col (B) within tile
    const int rowTile = wave >> 2;   // 0..1
    const int colTile = wave & 3;    // 0..3
    const int row0 = blockIdx.x * 32;
    const int col0 = blockIdx.y * 64;

    const int ar = tid >> 3, akq = (tid & 7) * 4;   // A slice: 8 B per thread
    const int bc = tid >> 2, bkq = (tid & 3) * 8;   // B slice: 16 B per thread
    const int agrow = row0 + ar;

    auto stage = [&](int buf, int kk) {
#if HAVE_ASYNC
        if (agrow < M)
            async_b64(&As[buf][ar][akq], A + (size_t)agrow * K + kk + akq);
        async_b128(&Bs[buf][bc][bkq], Bt + (size_t)(col0 + bc) * K + kk + bkq);
#else
        if (agrow < M)
            *(uint2*)&As[buf][ar][akq] = *(const uint2*)(A + (size_t)agrow * K + kk + akq);
        else
            *(uint2*)&As[buf][ar][akq] = make_uint2(0u, 0u);
        *(uint4*)&Bs[buf][bc][bkq] = *(const uint4*)(Bt + (size_t)(col0 + bc) * K + kk + bkq);
#endif
    };

    v8f acc = {};
    int cur = 0;
    stage(0, 0);
    for (int kk = 0; kk < K; kk += 32) {
        wait_async_lds();        // per-wave: own async stores done
        __syncthreads();         // cross-wave: everyone's stores visible
        if (kk + 32 < K) stage(cur ^ 1, kk + 32);   // prefetch next slab

        v16bf a, b;
#pragma unroll
        for (int j = 0; j < 8; ++j) {
            int k2 = (j < 4) ? (8 * g + 2 * j) : (16 + 8 * g + 2 * (j - 4));
            a[2 * j]     = As[cur][rowTile * 16 + rc][k2];
            a[2 * j + 1] = As[cur][rowTile * 16 + rc][k2 + 1];
            b[2 * j]     = Bs[cur][colTile * 16 + rc][k2];
            b[2 * j + 1] = Bs[cur][colTile * 16 + rc][k2 + 1];
        }
        acc = __builtin_amdgcn_wmma_f32_16x16x32_bf16(false, a, false, b,
                                                      (short)0, acc, false, false);
        __syncthreads();         // all reads of [cur] done before re-staging it
        cur ^= 1;
    }
#pragma unroll
    for (int r = 0; r < 8; ++r) {
        int grow = row0 + rowTile * 16 + g * 8 + r;
        int gcol = col0 + colTile * 16 + rc;
        if (grow < M) C[(size_t)grow * ncols + gcol] = acc[r];
    }
}

// ---------------------------------------------------------------------------
// EdgeConv edge kernel: per 16-edge tile,
//   act = ReLU(P[dst] + Q[src] + b1)  (bf16 in LDS),
//   m   = act @ W2 + b2  via WMMA (W2^T async-staged to LDS once per block),
//   segment-max into encOut[dst] using ordered-uint atomicMax.
// ---------------------------------------------------------------------------
__global__ void edgeconv_k(const int* __restrict__ src, const int* __restrict__ dst,
                           const float* __restrict__ PQ, const float* __restrict__ b1,
                           const __bf16* __restrict__ W2t, const float* __restrict__ b2,
                           unsigned* __restrict__ encOut, int E) {
    __shared__ __bf16 act[16][128];
    __shared__ __bf16 Ws[128][128];       // W2^T, [col][k]
    const int tid = threadIdx.x;
    const int e0  = blockIdx.x * 16;

    // stage whole W2^T (32 KB) once
#if HAVE_ASYNC
    for (int i = tid * 8; i < 128 * 128; i += 256 * 8)
        async_b128(&Ws[0][0] + i, W2t + i);
#else
    for (int i = tid * 8; i < 128 * 128; i += 256 * 8)
        *(uint4*)(&Ws[0][0] + i) = *(const uint4*)(W2t + i);
#endif

    for (int i = tid; i < 16 * 128; i += 256) {
        int r = i >> 7, c = i & 127;
        int e = e0 + r;
        float v = 0.0f;
        if (e < E) {
            int d = dst[e], s = src[e];
            v = PQ[(size_t)d * 256 + c] + PQ[(size_t)s * 256 + 128 + c] + b1[c];
            v = v > 0.0f ? v : 0.0f;
        }
        act[r][c] = (__bf16)v;
    }
    wait_async_lds();
    __syncthreads();

    const int lane = tid & 31, wave = tid >> 5;
    const int g = lane >> 4, rc = lane & 15;
    const int col = wave * 16 + rc;           // 8 waves cover 128 output cols
    v8f acc = {};
    for (int kk = 0; kk < 128; kk += 32) {
        v16bf a, b;
#pragma unroll
        for (int j = 0; j < 8; ++j) {
            int k2 = (j < 4) ? (8 * g + 2 * j) : (16 + 8 * g + 2 * (j - 4));
            a[2 * j]     = act[rc][kk + k2];
            a[2 * j + 1] = act[rc][kk + k2 + 1];
            b[2 * j]     = Ws[col][kk + k2];
            b[2 * j + 1] = Ws[col][kk + k2 + 1];
        }
        acc = __builtin_amdgcn_wmma_f32_16x16x32_bf16(false, a, false, b,
                                                      (short)0, acc, false, false);
    }
    float bias = b2[col];
#pragma unroll
    for (int r = 0; r < 8; ++r) {
        int e = e0 + g * 8 + r;
        if (e < E) {
            float v = acc[r] + bias;
            atomicMax(&encOut[(size_t)dst[e] * 128 + col], encodeOrdered(v));
        }
    }
}

__global__ void finalize_hec_k(const unsigned* __restrict__ enc,
                               float* __restrict__ hec, __bf16* __restrict__ hecbf,
                               long count) {
    long i = (long)blockIdx.x * 256 + threadIdx.x;
    if (i < count) {
        float f = decodeOrdered(enc[i]);
        hec[i] = f;
        hecbf[i] = (__bf16)f;
    }
}

// ---------------------------------------------------------------------------
// GAT pieces
// ---------------------------------------------------------------------------
__global__ void att_scores_k(const float* __restrict__ xp,
                             const float* __restrict__ attS, const float* __restrict__ attD,
                             float* __restrict__ asrc, float* __restrict__ adst, int n) {
    int i = blockIdx.x * 256 + threadIdx.x;       // over n*HEADS
    if (i >= n * 2) return;
    int node = i >> 1, h = i & 1;
    const float* xr = xp + (size_t)node * CC + h * HH;
    float s1 = 0.f, s2 = 0.f;
    for (int c = 0; c < HH; ++c) {
        float v = xr[c];
        s1 += v * attS[h * HH + c];
        s2 += v * attD[h * HH + c];
    }
    asrc[i] = s1; adst[i] = s2;
}

__global__ void gat_pass1_k(const int* __restrict__ src, const int* __restrict__ dst,
                            const float* __restrict__ asrc, const float* __restrict__ adst,
                            float* __restrict__ alpha, unsigned* __restrict__ amaxEnc,
                            int E, int n) {
    int e = blockIdx.x * 256 + threadIdx.x;
    if (e >= E + n) return;
    int s = (e < E) ? src[e] : (e - E);
    int d = (e < E) ? dst[e] : (e - E);
#pragma unroll
    for (int h = 0; h < 2; ++h) {
        float a = asrc[s * 2 + h] + adst[d * 2 + h];
        a = a > 0.f ? a : 0.2f * a;               // leaky_relu
        alpha[(size_t)e * 2 + h] = a;
        atomicMax(&amaxEnc[d * 2 + h], encodeOrdered(a));
    }
}

__global__ void decode_max_k(unsigned* __restrict__ buf, int count) {
    int i = blockIdx.x * 256 + threadIdx.x;
    if (i < count) {
        float f = decodeOrdered(buf[i]);
        ((float*)buf)[i] = f;
    }
}

__global__ void gat_pass2_k(const int* __restrict__ src, const int* __restrict__ dst,
                            const float* __restrict__ amax, float* __restrict__ alpha,
                            float* __restrict__ denom, int E, int n) {
    int e = blockIdx.x * 256 + threadIdx.x;
    if (e >= E + n) return;
    int d = (e < E) ? dst[e] : (e - E);
#pragma unroll
    for (int h = 0; h < 2; ++h) {
        float ex = __expf(alpha[(size_t)e * 2 + h] - amax[d * 2 + h]);
        alpha[(size_t)e * 2 + h] = ex;
        atomicAdd(&denom[d * 2 + h], ex);
    }
}

__global__ void gat_scatter_k(const int* __restrict__ src, const int* __restrict__ dst,
                              const float* __restrict__ xp, const float* __restrict__ exbuf,
                              const float* __restrict__ denom, float* __restrict__ agg,
                              int E, int n) {
    int gid = blockIdx.x * 256 + threadIdx.x;     // (E+n)*64 threads
    int e = gid >> 6;
    if (e >= E + n) return;
    int t  = gid & 63;
    int c0 = t * 4;                               // 4 cols per thread, head-uniform
    int h  = c0 >> 7;
    int s = (e < E) ? src[e] : (e - E);
    int d = (e < E) ? dst[e] : (e - E);
    float w = exbuf[(size_t)e * 2 + h] / denom[d * 2 + h];
    const float4 xv = *(const float4*)(xp + (size_t)s * CC + c0);
    float* ap = agg + (size_t)d * CC + c0;
    atomicAdd(ap + 0, xv.x * w);
    atomicAdd(ap + 1, xv.y * w);
    atomicAdd(ap + 2, xv.z * w);
    atomicAdd(ap + 3, xv.w * w);
}

__global__ void epilogue1_k(float* __restrict__ agg, const float* __restrict__ bias,
                            __bf16* __restrict__ hbf, long count) {
    long i = (long)blockIdx.x * 256 + threadIdx.x;
    if (i < count) {
        float v = agg[i] + bias[i & 255];
        v = v > 0.f ? v : 0.f;
        agg[i] = v;                                // h1 (fp32, residual source)
        hbf[i] = (__bf16)v;                        // h1 (bf16, GEMM input)
    }
}

__global__ void epilogue2_k(float* __restrict__ agg2, const float* __restrict__ bias,
                            const float* __restrict__ resid, long count) {
    long i = (long)blockIdx.x * 256 + threadIdx.x;
    if (i < count) {
        float v = agg2[i] + bias[i & 255];
        v = v > 0.f ? v : 0.f;
        agg2[i] = v + resid[i];                    // h2 in place
    }
}

__global__ void regressor_k(const float* __restrict__ h2, const float* __restrict__ W,
                            const float* __restrict__ b, float* __restrict__ out, int n) {
    int node = blockIdx.x * 256 + threadIdx.x;
    if (node >= n) return;
    const float* hr = h2 + (size_t)node * CC;
    float s = b[0];
    for (int c = 0; c < CC; ++c) s += hr[c] * W[c];
    out[node] = s;
}

// ---------------------------------------------------------------------------
// Host orchestration (graph-capture safe: kernels + hipMemsetAsync only)
// ---------------------------------------------------------------------------
extern "C" void kernel_launch(void* const* d_in, const int* in_sizes, int n_in,
                              void* d_out, int out_size, void* d_ws, size_t ws_size,
                              hipStream_t stream) {
    const float* x    = (const float*)d_in[0];
    const int*   ei   = (const int*)d_in[1];
    const float* ecW1 = (const float*)d_in[2];
    const float* ecb1 = (const float*)d_in[3];
    const float* ecW2 = (const float*)d_in[4];
    const float* ecb2 = (const float*)d_in[5];
    const float* g1W  = (const float*)d_in[6];
    const float* g1as = (const float*)d_in[7];
    const float* g1ad = (const float*)d_in[8];
    const float* g1b  = (const float*)d_in[9];
    const float* g2W  = (const float*)d_in[10];
    const float* g2as = (const float*)d_in[11];
    const float* g2ad = (const float*)d_in[12];
    const float* g2b  = (const float*)d_in[13];
    const float* regW = (const float*)d_in[14];
    const float* regb = (const float*)d_in[15];

    const int n = in_sizes[0] / 128;
    const int E = in_sizes[1] / 2;
    const int* src = ei;
    const int* dst = ei + E;

    char* w = (char*)d_ws;
    size_t off = 0;
    auto take = [&](size_t bytes) -> char* {
        char* p = w + off;
        off += (bytes + 255) & ~(size_t)255;
        return p;
    };
    __bf16*   xbf   = (__bf16*)take((size_t)n * 128 * 2);
    __bf16*   wct   = (__bf16*)take((size_t)256 * 128 * 2);
    __bf16*   w2t   = (__bf16*)take((size_t)128 * 128 * 2);
    __bf16*   g1t   = (__bf16*)take((size_t)256 * 128 * 2);
    __bf16*   g2t   = (__bf16*)take((size_t)256 * 256 * 2);
    float*    PQ    = (float*)take((size_t)n * 256 * 4);     // later reused as xp2
    unsigned* enc   = (unsigned*)take((size_t)n * 128 * 4);  // later reused as h1bf
    float*    hec   = (float*)take((size_t)n * 128 * 4);
    __bf16*   hecbf = (__bf16*)take((size_t)n * 128 * 2);
    float*    xp1   = (float*)take((size_t)n * 256 * 4);     // later reused as agg2
    float*    asrc  = (float*)take((size_t)n * 2 * 4);
    float*    adst  = (float*)take((size_t)n * 2 * 4);
    unsigned* amax  = (unsigned*)take((size_t)n * 2 * 4);    // decoded in place
    float*    denom = (float*)take((size_t)n * 2 * 4);
    float*    alpha = (float*)take((size_t)(E + n) * 2 * 4);
    float*    agg1  = (float*)take((size_t)n * 256 * 4);     // becomes h1 fp32

    __bf16* h1bf = (__bf16*)enc;
    float*  xp2  = PQ;
    float*  agg2 = xp1;

    const int T = 256;
    auto cdiv = [](long a, long b) { return (int)((a + b - 1) / b); };

    // ---- weight prep + x -> bf16 ----
    prep_weights_k<<<cdiv(147456, T), T, 0, stream>>>(ecW1, ecW2, g1W, g2W,
                                                      wct, w2t, g1t, g2t);
    convert_bf_k<<<cdiv((long)n * 128, T), T, 0, stream>>>(x, xbf, (long)n * 128);

    // ---- EdgeConv ----
    (void)hipMemsetAsync(enc, 0, (size_t)n * 128 * 4, stream);
    gemm_bf16_k<<<dim3(cdiv(n, 32), 4), T, 0, stream>>>(xbf, wct, PQ, n, 128, 256);
    edgeconv_k<<<cdiv(E, 16), T, 0, stream>>>(src, dst, PQ, ecb1, w2t, ecb2, enc, E);
    finalize_hec_k<<<cdiv((long)n * 128, T), T, 0, stream>>>(enc, hec, hecbf,
                                                             (long)n * 128);

    // ---- GAT layer 1 ----
    gemm_bf16_k<<<dim3(cdiv(n, 32), 4), T, 0, stream>>>(hecbf, g1t, xp1, n, 128, 256);
    att_scores_k<<<cdiv((long)n * 2, T), T, 0, stream>>>(xp1, g1as, g1ad, asrc, adst, n);
    (void)hipMemsetAsync(amax, 0, (size_t)n * 2 * 4, stream);
    (void)hipMemsetAsync(denom, 0, (size_t)n * 2 * 4, stream);
    (void)hipMemsetAsync(agg1, 0, (size_t)n * 256 * 4, stream);
    gat_pass1_k<<<cdiv((long)E + n, T), T, 0, stream>>>(src, dst, asrc, adst, alpha,
                                                        amax, E, n);
    decode_max_k<<<cdiv((long)n * 2, T), T, 0, stream>>>(amax, n * 2);
    gat_pass2_k<<<cdiv((long)E + n, T), T, 0, stream>>>(src, dst, (float*)amax, alpha,
                                                        denom, E, n);
    gat_scatter_k<<<cdiv(((long)E + n) * 64, T), T, 0, stream>>>(src, dst, xp1, alpha,
                                                                 denom, agg1, E, n);
    epilogue1_k<<<cdiv((long)n * 256, T), T, 0, stream>>>(agg1, g1b, h1bf,
                                                          (long)n * 256);

    // ---- GAT layer 2 ----
    gemm_bf16_k<<<dim3(cdiv(n, 32), 4), T, 0, stream>>>(h1bf, g2t, xp2, n, 256, 256);
    att_scores_k<<<cdiv((long)n * 2, T), T, 0, stream>>>(xp2, g2as, g2ad, asrc, adst, n);
    (void)hipMemsetAsync(amax, 0, (size_t)n * 2 * 4, stream);
    (void)hipMemsetAsync(denom, 0, (size_t)n * 2 * 4, stream);
    (void)hipMemsetAsync(agg2, 0, (size_t)n * 256 * 4, stream);
    gat_pass1_k<<<cdiv((long)E + n, T), T, 0, stream>>>(src, dst, asrc, adst, alpha,
                                                        amax, E, n);
    decode_max_k<<<cdiv((long)n * 2, T), T, 0, stream>>>(amax, n * 2);
    gat_pass2_k<<<cdiv((long)E + n, T), T, 0, stream>>>(src, dst, (float*)amax, alpha,
                                                        denom, E, n);
    gat_scatter_k<<<cdiv(((long)E + n) * 64, T), T, 0, stream>>>(src, dst, xp2, alpha,
                                                                 denom, agg2, E, n);
    epilogue2_k<<<cdiv((long)n * 256, T), T, 0, stream>>>(agg2, g2b, agg1,
                                                          (long)n * 256);

    // ---- regressor ----
    regressor_k<<<cdiv(n, T), T, 0, stream>>>(agg2, regW, regb, (float*)d_out, n);
}
